// GroupedGEMM_44367012168173
// MI455X (gfx1250) — matrix operand order repaired and checked by reading, compiled, not run on hardware
//
#include <hip/hip_runtime.h>

// ---------------------------------------------------------------------------
// Grouped GEMM (MoE expert GEMM), fp32 in/out, f16 WMMA compute w/ f32 accum.
// G=64 experts, tokens contiguous per expert, DIN=2560, DOUT=1664.
// Roofline: 69.8 GFLOP vs ~1.23 GB traffic (W=1.09GB dominant) -> memory
// bound at 23.3 TB/s (~53us) if matrix throughput >= ~1.3 PFLOP/s, which
// f16 WMMA easily clears (f32 WMMA would not). X (84MB) fits L2 (192MB) and
// is reused 13x -> keep RT; W/Y are streaming -> non-temporal hints.
// Double-buffered LDS -> single barrier per K-step.
// ---------------------------------------------------------------------------

typedef __attribute__((ext_vector_type(16))) _Float16 v16h;
typedef __attribute__((ext_vector_type(8)))  _Float16 h8;
typedef __attribute__((ext_vector_type(8)))  float    v8f;
typedef __attribute__((ext_vector_type(4)))  float    f4;

#define NG    64
#define NT    8192
#define DIN   2560
#define DOUT  1664
#define BM    128
#define BN    128
#define BK    32
#define LDR   40   // padded LDS row stride in halves (80B): 20 dwords/row,
                   // gcd(20,64)=4 -> 16-lane b128 frag loads hit all 64 banks

union Frag16 { v16h v; h8 h[2]; };
union Pack2  { _Float16 h[2]; unsigned int u; };

__global__ __launch_bounds__(256) void grouped_gemm_f16_wmma(
    const float* __restrict__ X,      // [T, DIN]
    const float* __restrict__ W,      // [G, DIN, DOUT]
    const int*   __restrict__ counts, // [G]
    float*       __restrict__ Y)      // [T, DOUT]
{
    // double-buffered staging tiles: 2 x (10 KB + 10 KB) = 40 KB
    __shared__ __align__(16) _Float16 sA[2][BM * LDR]; // [row][k], padded
    __shared__ __align__(16) _Float16 sB[2][BN * LDR]; // [n][k],  padded

    const int g     = blockIdx.x >> 1;     // expert id
    const int mtile = blockIdx.x & 1;      // which half of CAP=256
    const int n0    = blockIdx.y * BN;     // output-column tile

    // per-expert start offset (uniform scalar loop -> s_loads)
    int start = 0;
    for (int i = 0; i < g; ++i) start += counts[i];
    const int cnt = counts[g];
    const int m0  = mtile * BM;
    if (m0 >= cnt) return;                 // uniform early-exit

    const int tid  = threadIdx.x;
    const int lane = tid & 31;
    const int wave = tid >> 5;
    const int wm   = (wave >> 2) * 64;     // wave M origin: 0 or 64
    const int wn   = (wave & 3) * 32;      // wave N origin: 0,32,64,96
    const int lm   = lane & 15;
    const int lh   = lane >> 4;            // lane-half select

    // ---- X staging map: thread -> (row, k-half). 128 rows x 2 halves of 16.
    const int xr = tid >> 1;               // 0..127 local row
    const int xh = (tid & 1) * 16;         // 0 or 16 within BK
    long xrow = (long)start + m0 + xr;
    if (xrow >= NT) xrow = NT - 1;         // clamp; clamped rows never stored
    const float* __restrict__ xp = X + xrow * DIN;

    // ---- W staging map: wave w owns k rows 4w..4w+3; lane owns n = 4l..4l+3.
    //      -> coalesced 512B row reads; packed (k,k+1) b32 transpose stores.
    const int wkr = wave * 4;              // base k row of this wave
    const int wnc = lane * 4;              // base n col of this lane
    const float* __restrict__ wbase =
        W + ((long)g * DIN + wkr) * DOUT + n0 + wnc;

    v8f acc[4][2] = {};

    for (int kk = 0; kk < DIN / BK; ++kk) {
        const int k0  = kk * BK;
        const int buf = kk & 1;
        _Float16* __restrict__ tA = sA[buf];
        _Float16* __restrict__ tB = sB[buf];

        // ---------------- global loads into registers ----------------------
        f4 xv[4];
        #pragma unroll
        for (int q = 0; q < 4; ++q)
            xv[q] = *(const f4*)(xp + k0 + xh + q * 4);

        const float* wq = wbase + (long)k0 * DOUT;
        f4 wv[4];                           // wv[r] = row (k0+wkr+r), n = wnc..+3
        #pragma unroll
        for (int r = 0; r < 4; ++r)
            wv[r] = __builtin_nontemporal_load((const f4*)(wq + (long)r * DOUT));

        // prefetch next K-step (global_prefetch_b8)
        if (k0 + BK < DIN) {
            __builtin_prefetch(xp + k0 + BK + xh, 0, 3);
            __builtin_prefetch(wq + (long)BK * DOUT, 0, 3);
        }

        // ---------------- stage to LDS, converting f32 -> f16 ---------------
        // Safe without a pre-store barrier: buf was last read at iteration
        // kk-2, and every wave passed the kk-1 barrier after those reads.
        {   // A: two 16B packed stores per thread
            h8 h0, h1;
            #pragma unroll
            for (int u = 0; u < 4; ++u) {
                h0[u]     = (_Float16)xv[0][u];
                h0[u + 4] = (_Float16)xv[1][u];
                h1[u]     = (_Float16)xv[2][u];
                h1[u + 4] = (_Float16)xv[3][u];
            }
            *(h8*)&tA[xr * LDR + xh]     = h0;
            *(h8*)&tA[xr * LDR + xh + 8] = h1;
        }
        {   // B: transpose 4k x 4n register micro-tile -> packed b32 stores
            #pragma unroll
            for (int j = 0; j < 4; ++j) {
                const int n = wnc + j;
                Pack2 p01, p23;
                p01.h[0] = (_Float16)wv[0][j];
                p01.h[1] = (_Float16)wv[1][j];
                p23.h[0] = (_Float16)wv[2][j];
                p23.h[1] = (_Float16)wv[3][j];
                *(unsigned int*)&tB[n * LDR + wkr]     = p01.u;
                *(unsigned int*)&tB[n * LDR + wkr + 2] = p23.u;
            }
        }

        __syncthreads();  // single barrier per K-step: staging visible

        // ---------------- fragment loads per ISA VGPR layout ----------------
        // A (16x32 f16): lanes 0-15 hold row M=lm with K in [0..7]+[16..23],
        //                lanes 16-31 the K in [8..15]+[24..31] halves.
        Frag16 a[4];
        #pragma unroll
        for (int i = 0; i < 4; ++i) {
            const _Float16* p = &tA[(wm + i * 16 + lm) * LDR + lh * 8];
            a[i].h[0] = *(const h8*)(p);
            a[i].h[1] = *(const h8*)(p + 16);
        }
        // B (32x16 f16): lane holds column N=lm; lanes 0-15: K in [0..7]+[16..23],
        //                lanes 16-31: K in [8..15]+[24..31] (matches A).
        Frag16 b[2];
        #pragma unroll
        for (int j = 0; j < 2; ++j) {
            const _Float16* p = &tB[(wn + j * 16 + lm) * LDR + lh * 8];
            b[j].h[0] = *(const h8*)(p);
            b[j].h[1] = *(const h8*)(p + 16);
        }

        // ---------------- 8x v_wmma_f32_16x16x32_f16 ------------------------
        #pragma unroll
        for (int i = 0; i < 4; ++i) {
            #pragma unroll
            for (int j = 0; j < 2; ++j) {
                acc[i][j] = __builtin_amdgcn_wmma_f32_16x16x32_f16(
                    /*neg_a=*/false, a[i].v,
                    /*neg_b=*/false, b[j].v,
                    /*c_mod=*/(short)0, acc[i][j],
                    /*reuse_a=*/false, /*reuse_b=*/false);
            }
        }
    }

    // ---------------- store C fragments (predicated on ragged count) -------
    // C layout: lane = column (N=lm), VGPR r = row (lanes 0-15: M=r,
    //           lanes 16-31: M=8+r). Y is write-once -> non-temporal.
    #pragma unroll
    for (int i = 0; i < 4; ++i) {
        const int rbase = m0 + wm + i * 16 + lh * 8; // local row within expert
        #pragma unroll
        for (int j = 0; j < 2; ++j) {
            const int col = n0 + wn + j * 16 + lm;
            #pragma unroll
            for (int r = 0; r < 8; ++r) {
                const int rloc = rbase + r;
                if (rloc < cnt)
                    __builtin_nontemporal_store(
                        acc[i][j][r], &Y[((long)start + rloc) * DOUT + col]);
            }
        }
    }
}

extern "C" void kernel_launch(void* const* d_in, const int* in_sizes, int n_in,
                              void* d_out, int out_size, void* d_ws, size_t ws_size,
                              hipStream_t stream) {
    const float* X      = (const float*)d_in[0];   // [T, DIN]  fp32
    const float* W      = (const float*)d_in[1];   // [G, DIN, DOUT] fp32
    const int*   counts = (const int*)d_in[2];     // [G] int32
    float*       Y      = (float*)d_out;           // [T, DOUT] fp32

    dim3 grid(NG * 2, DOUT / BN);   // 128 m-tiles x 13 n-tiles
    grouped_gemm_f16_wmma<<<grid, 256, 0, stream>>>(X, W, counts, Y);
}
